// T5Attention_72550587564155
// MI455X (gfx1250) — matrix operand order repaired and checked by reading, not compiled, not run on hardware
//
#include <hip/hip_runtime.h>
#include <hip/hip_bf16.h>

// ---------------------------------------------------------------------------
// T5 attention for MI455X (gfx1250): bf16 WMMA 16x16x32, fp32 accumulate.
// B=2, L=2048, DIM=1024, NUM_HEADS=16, HEAD_DIM=64.
// ---------------------------------------------------------------------------

#define LSEQ 2048
#define DIMV 1024
#define NHEAD 16
#define HDIM 64
#define NEG_MAX (-3.4028234663852886e38f)

typedef __attribute__((ext_vector_type(16))) __bf16 v16bf;
typedef __attribute__((ext_vector_type(8)))  float  v8f;

union FragBF {            // one WMMA 16-bf16 operand = 8 VGPRs = 32 bytes
    v16bf v;
    uint4 q[2];
    unsigned short u[16];
};

__device__ __forceinline__ unsigned short f2bf(float f) {
    union { float f; unsigned u; } c; c.f = f;
    unsigned u = c.u;
    unsigned r = u + 0x7FFFu + ((u >> 16) & 1u);   // round-to-nearest-even
    return (unsigned short)(r >> 16);
}

__device__ __forceinline__ v8f wmma_bf16(const FragBF& a, const FragBF& b, v8f c) {
    return __builtin_amdgcn_wmma_f32_16x16x32_bf16(
        /*neg_a=*/false, a.v, /*neg_b=*/false, b.v,
        /*c_mod=*/(short)0, c, /*reuse_a=*/false, /*reuse_b=*/false);
}

// XOR lane swizzle inside 32-lane groups: ds_swizzle_b32 offset =
// {and_mask=0x1f, or_mask=0, xor_mask}; masks 1/2/4/8 stay within 16-lane halves.
template <int XORMASK>
__device__ __forceinline__ float lane_xor_f32(float x) {
    union { float f; int i; } c; c.f = x;
    c.i = __builtin_amdgcn_ds_swizzle(c.i, 0x7C00 | XORMASK);
    return c.f;
}

__device__ __forceinline__ float half_reduce_max(float v) {
    v = fmaxf(v, lane_xor_f32<1>(v));
    v = fmaxf(v, lane_xor_f32<2>(v));
    v = fmaxf(v, lane_xor_f32<4>(v));
    v = fmaxf(v, lane_xor_f32<8>(v));
    return v;
}
__device__ __forceinline__ float half_reduce_sum(float v) {
    v += lane_xor_f32<1>(v);
    v += lane_xor_f32<2>(v);
    v += lane_xor_f32<4>(v);
    v += lane_xor_f32<8>(v);
    return v;
}

// ---------------------------------------------------------------------------
// fp32 -> bf16, 4 elements/thread (n multiple of 4)
// ---------------------------------------------------------------------------
__global__ void cvt_bf16_kernel(const float* __restrict__ src,
                                unsigned short* __restrict__ dst, int n) {
    int i = (blockIdx.x * blockDim.x + threadIdx.x) * 4;
    if (i < n) {
        float4 f = *(const float4*)(src + i);
        unsigned lo = (unsigned)f2bf(f.x) | ((unsigned)f2bf(f.y) << 16);
        unsigned hi = (unsigned)f2bf(f.z) | ((unsigned)f2bf(f.w) << 16);
        uint2 o; o.x = lo; o.y = hi;
        *(uint2*)(dst + i) = o;
    }
}

// W[K][N] fp32 -> WT[N][K] bf16  (weights are small: 4MB total traffic)
__global__ void cvt_transpose_kernel(const float* __restrict__ W,
                                     unsigned short* __restrict__ WT,
                                     int K, int N) {
    int idx = blockIdx.x * blockDim.x + threadIdx.x;
    if (idx < N * K) {
        int n = idx / K;
        int k = idx - n * K;
        WT[idx] = f2bf(W[(size_t)k * N + n]);
    }
}

// ---------------------------------------------------------------------------
// C[M][N] = A[M][K] * BT[N][K]^T   (bf16 inputs, fp32 accumulate)
// 8 waves / block; each wave owns a 32x32 output tile (2x2 WMMA register tile:
// 2 A-frags + 2 B-frags -> 4 WMMAs per 32-deep K step, 16 flops/byte).
// STORE_MODE: 0 = fp32 row-major, 1 = bf16 row-major,
//             2 = bf16 transposed per batch: Vt[b][col][l], l = row % 2048
// ---------------------------------------------------------------------------
template <int STORE_MODE>
__global__ __launch_bounds__(256) void gemm_bf16_kernel(
    const unsigned short* __restrict__ A,
    const unsigned short* __restrict__ BT,
    void* __restrict__ Cv, int M, int N, int K) {
    const int wid  = threadIdx.x >> 5;
    const int lane = threadIdx.x & 31;
    const int lid  = lane & 15;
    const int half = lane >> 4;

    const int tiles_n = N >> 5;                       // 32-wide tiles
    const int tile = blockIdx.x * 8 + wid;
    if (tile >= (M >> 5) * tiles_n) return;           // uniform per wave
    const int tm = tile / tiles_n;
    const int tn = tile - tm * tiles_n;

    const unsigned short* arow0 = A  + (size_t)(tm * 32 + lid) * K;        // lane = M
    const unsigned short* arow1 = arow0 + (size_t)16 * K;
    const unsigned short* brow0 = BT + (size_t)(tn * 32 + lid) * K;        // lane = N
    const unsigned short* brow1 = brow0 + (size_t)16 * K;

    v8f acc00, acc01, acc10, acc11;
#pragma unroll
    for (int r = 0; r < 8; ++r) { acc00[r] = 0.f; acc01[r] = 0.f;
                                  acc10[r] = 0.f; acc11[r] = 0.f; }

    for (int k0 = 0; k0 < K; k0 += 32) {
        FragBF a0, a1, b0, b1;
        // A 16x32: lane row = M, K = half*16 + i (contiguous in A row)
        a0.q[0] = *(const uint4*)(arow0 + k0 + half * 16);
        a0.q[1] = *(const uint4*)(arow0 + k0 + half * 16 + 8);
        a1.q[0] = *(const uint4*)(arow1 + k0 + half * 16);
        a1.q[1] = *(const uint4*)(arow1 + k0 + half * 16 + 8);
        // B 32x16: lane col = N, K = half*16 + i (contiguous in BT row)
        b0.q[0] = *(const uint4*)(brow0 + k0 + half * 16);
        b0.q[1] = *(const uint4*)(brow0 + k0 + half * 16 + 8);
        b1.q[0] = *(const uint4*)(brow1 + k0 + half * 16);
        b1.q[1] = *(const uint4*)(brow1 + k0 + half * 16 + 8);
        // keep the streams ahead of the matrix pipe
        __builtin_prefetch(arow0 + k0 + 32, 0, 0);
        __builtin_prefetch(arow1 + k0 + 32, 0, 0);
        __builtin_prefetch(brow0 + k0 + 32, 0, 0);
        __builtin_prefetch(brow1 + k0 + 32, 0, 0);

        acc00 = wmma_bf16(a0, b0, acc00);
        acc01 = wmma_bf16(a0, b1, acc01);
        acc10 = wmma_bf16(a1, b0, acc10);
        acc11 = wmma_bf16(a1, b1, acc11);
    }

    // C/D frag: VGPR r -> M = half*8 + r (within 16-row subtile), lane -> N
    const int col0 = tn * 32 + lid;
#pragma unroll
    for (int sm = 0; sm < 2; ++sm) {
#pragma unroll
        for (int sn = 0; sn < 2; ++sn) {
            const v8f& acc = sm == 0 ? (sn == 0 ? acc00 : acc01)
                                     : (sn == 0 ? acc10 : acc11);
            const int row0 = tm * 32 + sm * 16 + half * 8;
            const int col  = col0 + sn * 16;
            if (STORE_MODE == 0) {
                float* C = (float*)Cv;
#pragma unroll
                for (int r = 0; r < 8; ++r)
                    C[(size_t)(row0 + r) * N + col] = acc[r];
            } else if (STORE_MODE == 1) {
                unsigned short* C = (unsigned short*)Cv;
#pragma unroll
                for (int r = 0; r < 8; ++r)
                    C[(size_t)(row0 + r) * N + col] = f2bf(acc[r]);
            } else {
                unsigned short* C = (unsigned short*)Cv;   // Vt[b][N][2048]
#pragma unroll
                for (int r = 0; r < 8; ++r) {
                    int m = row0 + r;
                    int bb = m >> 11;          // batch
                    int l  = m & 2047;         // sequence pos
                    C[((size_t)bb * N + col) * LSEQ + l] = f2bf(acc[r]);
                }
            }
        }
    }
}

// ---------------------------------------------------------------------------
// Flash attention: one wave per (b, head, 16-query tile). 32 keys / iteration.
//  S(16x32) = Q(16x64) * K^T      4 WMMAs
//  online softmax (+pos_bias, mask -> -FLT_MAX, branchless cndmask select)
//  O(16x64) += P(16x32) * V       4 WMMAs, P re-laid out C-frag -> A-frag via LDS
// All per-row state loops force-unrolled so mrow/lsum/o* stay in static VGPRs
// (no M0/v_movrels indexed register file access).
// ---------------------------------------------------------------------------
__global__ __launch_bounds__(32) void attn_kernel(
    const unsigned short* __restrict__ Qb,   // [B][L][1024] bf16
    const unsigned short* __restrict__ Kb,   // [B][L][1024] bf16
    const unsigned short* __restrict__ Vt,   // [B][1024][L] bf16 (transposed)
    const float* __restrict__ pos_bias,      // [NHEAD][L][L] fp32
    const int*   __restrict__ mask,          // [B][L]
    unsigned short* __restrict__ Ob) {       // [B][L][1024] bf16
    const int lane = threadIdx.x;
    const int lid  = lane & 15;
    const int half = lane >> 4;
    const int q0   = blockIdx.x * 16;
    const int h    = blockIdx.y;
    const int b    = blockIdx.z;

    __shared__ unsigned short Plds[16 * 32];

    // Q fragments (rows q0+lid, head dims 0..63): two A-frags, loaded once
    const unsigned short* qrow = Qb + ((size_t)(b * LSEQ + q0 + lid) * DIMV + h * HDIM);
    FragBF qa0, qa1;
    qa0.q[0] = *(const uint4*)(qrow +      half * 16);
    qa0.q[1] = *(const uint4*)(qrow +      half * 16 + 8);
    qa1.q[0] = *(const uint4*)(qrow + 32 + half * 16);
    qa1.q[1] = *(const uint4*)(qrow + 32 + half * 16 + 8);

    v8f o0, o1, o2, o3;
    float mrow[8], lsum[8];
#pragma unroll
    for (int r = 0; r < 8; ++r) {
        o0[r] = 0.f; o1[r] = 0.f; o2[r] = 0.f; o3[r] = 0.f;
        mrow[r] = NEG_MAX; lsum[r] = 0.f;
    }

    const float* pbrow = pos_bias + ((size_t)h * LSEQ + q0 + half * 8) * LSEQ;

    for (int kc = 0; kc < LSEQ / 32; ++kc) {
        const int key0 = kc * 32;

        const unsigned short* kr0 =
            Kb + ((size_t)(b * LSEQ + key0 + lid) * DIMV + h * HDIM);
        const unsigned short* kr1 = kr0 + (size_t)16 * DIMV;
        const unsigned short* vcol =
            Vt + ((size_t)(b * DIMV + h * HDIM + lid) * LSEQ + key0 + half * 16);

        // prefetch the next key-chunk streams (speculative, dropped if invalid)
        __builtin_prefetch(kr0 + (size_t)32 * DIMV, 0, 0);
        __builtin_prefetch(kr1 + (size_t)32 * DIMV, 0, 0);
        __builtin_prefetch(vcol + 32, 0, 0);
        __builtin_prefetch(pbrow + key0 + 32, 0, 0);

        // ---- S = Q * K^T over 32 keys (two 16-key subtiles) ----
        v8f s0, s1;
#pragma unroll
        for (int r = 0; r < 8; ++r) { s0[r] = 0.f; s1[r] = 0.f; }
        {
            FragBF kb;
            kb.q[0] = *(const uint4*)(kr0 + half * 16);
            kb.q[1] = *(const uint4*)(kr0 + half * 16 + 8);
            s0 = wmma_bf16(qa0, kb, s0);
            kb.q[0] = *(const uint4*)(kr0 + 32 + half * 16);
            kb.q[1] = *(const uint4*)(kr0 + 32 + half * 16 + 8);
            s0 = wmma_bf16(qa1, kb, s0);
            kb.q[0] = *(const uint4*)(kr1 + half * 16);
            kb.q[1] = *(const uint4*)(kr1 + half * 16 + 8);
            s1 = wmma_bf16(qa0, kb, s1);
            kb.q[0] = *(const uint4*)(kr1 + 32 + half * 16);
            kb.q[1] = *(const uint4*)(kr1 + 32 + half * 16 + 8);
            s1 = wmma_bf16(qa1, kb, s1);
        }

        // ---- additive bias: UNCONDITIONAL loads, then v_cndmask select ----
        const int kg0 = key0 + lid;
        const int kg1 = key0 + 16 + lid;
        const bool ok0 = mask[b * LSEQ + kg0] > 0;
        const bool ok1 = mask[b * LSEQ + kg1] > 0;
        float pv0[8], pv1[8];
#pragma unroll
        for (int r = 0; r < 8; ++r) {
            pv0[r] = pbrow[(size_t)r * LSEQ + kg0];    // always valid address
            pv1[r] = pbrow[(size_t)r * LSEQ + kg1];
        }
#pragma unroll
        for (int r = 0; r < 8; ++r) {
            s0[r] += ok0 ? pv0[r] : NEG_MAX;
            s1[r] += ok1 ? pv1[r] : NEG_MAX;
        }

        // ---- online softmax; write P (bf16) to LDS ----
#pragma unroll
        for (int r = 0; r < 8; ++r) {
            const float v    = half_reduce_max(fmaxf(s0[r], s1[r]));
            const float mnew = fmaxf(mrow[r], v);
            const float sc   = __expf(mrow[r] - mnew);
            mrow[r] = mnew;
            const float p0 = __expf(s0[r] - mnew);
            const float p1 = __expf(s1[r] - mnew);
            const float rs = half_reduce_sum(p0 + p1);
            lsum[r] = lsum[r] * sc + rs;
            o0[r] *= sc; o1[r] *= sc; o2[r] *= sc; o3[r] *= sc;
            const int row = half * 8 + r;       // C-frag ownership -> [row][key] LDS
            Plds[row * 32 + lid]      = f2bf(p0);
            Plds[row * 32 + 16 + lid] = f2bf(p1);
        }
        __syncthreads();

        // ---- P as A-fragment (16x32) ----
        FragBF pa;
        pa.q[0] = *(const uint4*)&Plds[lid * 32 + half * 16];
        pa.q[1] = *(const uint4*)&Plds[lid * 32 + half * 16 + 8];

        // ---- O += P * V  (V transposed: contiguous along keys) ----
        FragBF vb;
        vb.q[0] = *(const uint4*)(vcol);
        vb.q[1] = *(const uint4*)(vcol + 8);
        o0 = wmma_bf16(pa, vb, o0);
        vb.q[0] = *(const uint4*)(vcol + 16 * LSEQ);
        vb.q[1] = *(const uint4*)(vcol + 16 * LSEQ + 8);
        o1 = wmma_bf16(pa, vb, o1);
        vb.q[0] = *(const uint4*)(vcol + 32 * LSEQ);
        vb.q[1] = *(const uint4*)(vcol + 32 * LSEQ + 8);
        o2 = wmma_bf16(pa, vb, o2);
        vb.q[0] = *(const uint4*)(vcol + 48 * LSEQ);
        vb.q[1] = *(const uint4*)(vcol + 48 * LSEQ + 8);
        o3 = wmma_bf16(pa, vb, o3);
        __syncthreads();
    }

    // ---- normalize + store O (bf16) ----
#pragma unroll
    for (int r = 0; r < 8; ++r) {
        const float inv = 1.0f / lsum[r];
        const size_t base =
            (size_t)(b * LSEQ + q0 + half * 8 + r) * DIMV + h * HDIM + lid;
        Ob[base +  0] = f2bf(o0[r] * inv);
        Ob[base + 16] = f2bf(o1[r] * inv);
        Ob[base + 32] = f2bf(o2[r] * inv);
        Ob[base + 48] = f2bf(o3[r] * inv);
    }
}

// ---------------------------------------------------------------------------
// Host launcher
// ---------------------------------------------------------------------------
extern "C" void kernel_launch(void* const* d_in, const int* in_sizes, int n_in,
                              void* d_out, int out_size, void* d_ws, size_t ws_size,
                              hipStream_t stream) {
    const float* x        = (const float*)d_in[0];
    const int*   mask     = (const int*)  d_in[1];
    const float* pos_bias = (const float*)d_in[2];
    const float* Wq       = (const float*)d_in[3];
    const float* Wk       = (const float*)d_in[4];
    const float* Wv       = (const float*)d_in[5];
    const float* Wo       = (const float*)d_in[6];
    float* out = (float*)d_out;

    const int NX = 2 * LSEQ * DIMV;     // 4,194,304 elements in x / Q / K / V / O
    const int NW = DIMV * DIMV;         // 1,048,576 per weight matrix

    unsigned short* ws  = (unsigned short*)d_ws;
    unsigned short* xbf = ws;                 // bf16 x
    unsigned short* WqT = xbf + NX;           // bf16 Wq^T [N][K]
    unsigned short* WkT = WqT + NW;
    unsigned short* WvT = WkT + NW;
    unsigned short* WoT = WvT + NW;
    unsigned short* Qb  = WoT + NW;           // bf16 Q  [B][L][1024]
    unsigned short* Kb  = Qb  + NX;           // bf16 K  [B][L][1024]
    unsigned short* Vt  = Kb  + NX;           // bf16 V^T [B][1024][L]
    unsigned short* Ob  = Vt  + NX;           // bf16 attention output

    // 1) precision conversion (+ weight transposes)
    cvt_bf16_kernel<<<(NX / 4 + 255) / 256, 256, 0, stream>>>(x, xbf, NX);
    cvt_transpose_kernel<<<(NW + 255) / 256, 256, 0, stream>>>(Wq, WqT, DIMV, DIMV);
    cvt_transpose_kernel<<<(NW + 255) / 256, 256, 0, stream>>>(Wk, WkT, DIMV, DIMV);
    cvt_transpose_kernel<<<(NW + 255) / 256, 256, 0, stream>>>(Wv, WvT, DIMV, DIMV);
    cvt_transpose_kernel<<<(NW + 255) / 256, 256, 0, stream>>>(Wo, WoT, DIMV, DIMV);

    // 2) projections: (4096 x 1024) @ (1024 x 1024); 4096 32x32 tiles / 8 waves
    const int M = 2 * LSEQ, N = DIMV, K = DIMV;
    const int gemm_blocks = (M / 32) * (N / 32) / 8;
    gemm_bf16_kernel<1><<<gemm_blocks, 256, 0, stream>>>(xbf, WqT, Qb, M, N, K);
    gemm_bf16_kernel<1><<<gemm_blocks, 256, 0, stream>>>(xbf, WkT, Kb, M, N, K);
    gemm_bf16_kernel<2><<<gemm_blocks, 256, 0, stream>>>(xbf, WvT, Vt, M, N, K);

    // 3) flash attention: one wave per (b, head, 16-query tile)
    attn_kernel<<<dim3(LSEQ / 16, NHEAD, 2), 32, 0, stream>>>(
        Qb, Kb, Vt, pos_bias, mask, Ob);

    // 4) output projection -> fp32 d_out
    gemm_bf16_kernel<0><<<gemm_blocks, 256, 0, stream>>>(Ob, WoT, out, M, N, K);
}
